// TriangleAttention_40200893891142
// MI455X (gfx1250) — matrix operand order repaired and checked by reading, not compiled, not run on hardware
//
#include <hip/hip_runtime.h>
#include <hip/hip_bf16.h>

typedef __bf16 bf16;
typedef __attribute__((ext_vector_type(16))) __bf16 v16bf;
typedef __attribute__((ext_vector_type(8)))  float  v8f;

#define NSEQ 256
#define CZ   128
#define HC   128           // H * c = 4 * 32
#define WT_LD 132          // padded LDS stride for transposed weights
#define VT_LD 272          // padded LDS stride for transposed V (8-bank skew)

// ---------------- WMMA fragment loaders (CDNA5 16x16x32 bf16 layouts) -------

// A fragment: 16x32 (MxK), base points at tile origin, ld = row stride (elems).
// lane<16: row=lane, K in {0..7, 16..23}; lane>=16: row=lane-16, K in {8..15, 24..31}
__device__ __forceinline__ v16bf load_a(const bf16* base, int ld, int lane) {
    int r = lane & 15;
    int khalf = (lane >> 4) & 1;
    const bf16* p = base + r * ld + khalf * 8;
    v16bf a;
#pragma unroll
    for (int t = 0; t < 8; ++t) { a[t] = p[t]; a[t + 8] = p[16 + t]; }
    return a;
}

// B fragment 32x16 (KxN) where logical B[k][n] = src[n][k] (src row-major, stride lds)
// lane = column n (mod 16); lane<16 holds K in {0..7, 16..23}, lane>=16 holds K in {8..15, 24..31}
__device__ __forceinline__ v16bf load_bT(const bf16* src, int lds, int lane) {
    int n = lane & 15;
    int k0 = ((lane >> 4) & 1) * 8;
    const bf16* p = src + n * lds + k0;
    v16bf b;
#pragma unroll
    for (int t = 0; t < 8; ++t) { b[t] = p[t]; b[t + 8] = p[16 + t]; }
    return b;
}

#define WMMA_BF16(a, b, c) \
    __builtin_amdgcn_wmma_f32_16x16x32_bf16(false, (a), false, (b), (short)0, (c), false, false)

// Async global->LDS copy, 16 bytes per lane (GLOBAL_LOAD_ASYNC_TO_LDS_B128, GV mode).
// VGPR dest operand holds the LDS byte address (low 32 bits of the generic pointer).
__device__ __forceinline__ void async_cp16(void* lds_dst, const void* gsrc) {
    unsigned ldsoff = (unsigned)(unsigned long long)(uintptr_t)lds_dst;
    asm volatile("global_load_async_to_lds_b128 %0, %1, off"
                 :: "v"(ldsoff), "v"(gsrc)
                 : "memory");
}
__device__ __forceinline__ void async_wait0() {
    asm volatile("s_wait_asynccnt 0x0" ::: "memory");
}

// ---------------- Kernel 1: LayerNorm + q/k/v/g projection + pair bias ------
__global__ __launch_bounds__(256) void k1_ln_proj(
    const float* __restrict__ z, const float* __restrict__ ln_g, const float* __restrict__ ln_b,
    const float* __restrict__ Wq, const float* __restrict__ Wk, const float* __restrict__ Wv,
    const float* __restrict__ Wb, const float* __restrict__ Wg,
    bf16* __restrict__ qb, bf16* __restrict__ kb, bf16* __restrict__ vb, bf16* __restrict__ gb,
    float* __restrict__ biasb)
{
    extern __shared__ char smem[];
    bf16* A  = (bf16*)smem;                      // 64 x 128 layernormed rows (bf16)
    bf16* Wt = (bf16*)(smem + 64 * CZ * 2);      // 128 x WT_LD transposed weights

    const int tid  = threadIdx.x;
    const int lane = tid & 31;
    const int wave = tid >> 5;
    const long rows0 = (long)blockIdx.x * 64;    // flattened (j,k) row index

    // --- layernorm: 4 lanes per row, 32 elements each ---
    {
        int r = tid >> 2, part = tid & 3;
        const float* zr = z + (rows0 + r) * CZ + part * 32;
        float x[32];
        float s = 0.f, s2 = 0.f;
#pragma unroll
        for (int u = 0; u < 32; ++u) { x[u] = zr[u]; s += x[u]; s2 += x[u] * x[u]; }
        s  += __shfl_xor(s, 1, 32);  s  += __shfl_xor(s, 2, 32);
        s2 += __shfl_xor(s2, 1, 32); s2 += __shfl_xor(s2, 2, 32);
        float mu  = s * (1.f / 128.f);
        float var = s2 * (1.f / 128.f) - mu * mu;
        float inv = rsqrtf(var + 1e-5f);
#pragma unroll
        for (int u = 0; u < 32; ++u) {
            int c = part * 32 + u;
            float v = (x[u] - mu) * inv * ln_g[c] + ln_b[c];
            A[r * CZ + c] = (bf16)v;
        }
    }
    __syncthreads();

    // --- pair bias: one thread per (row, head) ---
    {
        int r = tid >> 2, h = tid & 3;
        float acc = 0.f;
#pragma unroll 4
        for (int c = 0; c < CZ; ++c) acc += (float)A[r * CZ + c] * Wb[c * 4 + h];
        biasb[(size_t)h * (NSEQ * NSEQ) + rows0 + r] = acc;
    }

    // --- four GEMMs: [64 x 128] x [128 x 128] via WMMA ---
    const float* Ws[4] = {Wq, Wk, Wv, Wg};
    bf16*        Ds[4] = {qb, kb, vb, gb};
#pragma unroll
    for (int w = 0; w < 4; ++w) {
        __syncthreads();
        // load weights transposed into LDS: Wt[n][c]
        for (int u = 0; u < 64; ++u) {
            int e = u * 256 + tid;          // 0 .. 16383
            int c = e >> 7, n = e & 127;
            Wt[n * WT_LD + c] = (bf16)Ws[w][e];
        }
        __syncthreads();

        int m  = wave & 3;                  // m-tile (row block of 16)
        int nb = (wave >> 2) * 4;           // first of 4 n-tiles
        v8f acc[4] = {};
#pragma unroll
        for (int kk = 0; kk < 4; ++kk) {
            v16bf a = load_a(A + m * 16 * CZ + kk * 32, CZ, lane);
#pragma unroll
            for (int t = 0; t < 4; ++t) {
                v16bf b = load_bT(Wt + (nb + t) * 16 * WT_LD + kk * 32, WT_LD, lane);
                acc[t] = WMMA_BF16(a, b, acc[t]);
            }
        }
        int n = lane & 15, hi = lane >> 4;
#pragma unroll
        for (int t = 0; t < 4; ++t) {
#pragma unroll
            for (int r = 0; r < 8; ++r) {
                float vv = acc[t][r];
                if (w == 0) vv *= 0.17677669529663687f;     // q pre-scaled by 1/sqrt(c)
                if (w == 3) vv = 1.f / (1.f + __expf(-vv)); // sigmoid gate
                long row = rows0 + m * 16 + hi * 8 + r;
                Ds[w][row * HC + (nb + t) * 16 + n] = (bf16)vv;
            }
        }
    }
}

// ---------------- Kernel 2: attention per (i, h, 64-query tile) -------------
__global__ __launch_bounds__(256) void k2_attn(
    const bf16* __restrict__ qb, const bf16* __restrict__ kb, const bf16* __restrict__ vb,
    const float* __restrict__ biasb, bf16* __restrict__ ob)
{
    extern __shared__ char smem[];
    float* S  = (float*)smem;                                   // 64 x 256 f32
    bf16*  P  = (bf16*)(smem + 64 * 256 * 4);                   // 64 x 256 bf16
    bf16*  Q  = (bf16*)(smem + 64 * 256 * 4 + 64 * 256 * 2);    // 64 x 32
    bf16*  K  = Q + 64 * 32;                                    // 256 x 32
    bf16*  V  = K + 256 * 32;                                   // 256 x 32 (row-major staging)
    bf16*  Vt = V + 256 * 32;                                   // 32 x VT_LD (transposed)

    const int tid  = threadIdx.x;
    const int lane = tid & 31;
    const int wave = tid >> 5;
    const int jt = blockIdx.x, h = blockIdx.y, i = blockIdx.z;
    const int j0 = jt * 64;
    const size_t base = (size_t)i * NSEQ * HC + (size_t)h * 32;

    // --- async copy Q (64x32), K (256x32), V (256x32) straight into LDS ---
    {
        int r = tid >> 2, part = tid & 3;
        async_cp16(Q + r * 32 + part * 8, qb + base + (size_t)(j0 + r) * HC + part * 8);
    }
#pragma unroll
    for (int u = 0; u < 4; ++u) {
        int idx = u * 256 + tid;            // 16B chunk index, 0..1023
        int r = idx >> 2, part = idx & 3;
        async_cp16(K + r * 32 + part * 8, kb + base + (size_t)r * HC + part * 8);
        async_cp16(V + r * 32 + part * 8, vb + base + (size_t)r * HC + part * 8);
    }
    async_wait0();
    __syncthreads();

    // --- transpose V into Vt[d][key] so P*V B-fragments are contiguous ---
    {
#pragma unroll 8
        for (int u = 0; u < 32; ++u) Vt[u * VT_LD + tid] = V[tid * 32 + u];
    }

    // S = Q K^T + bias  (q already scaled); 8 waves x 8 tiles of 16x16, k = 32 = c
    {
        int mj  = wave & 3;
        int nk0 = (wave >> 2) * 8;
        int n = lane & 15, hi = lane >> 4;
        v16bf a = load_a(Q + mj * 16 * 32, 32, lane);
#pragma unroll
        for (int t = 0; t < 8; ++t) {
            int nk = nk0 + t;
            v16bf b = load_bT(K + nk * 16 * 32, 32, lane);
            v8f acc = {};
            acc = WMMA_BF16(a, b, acc);
#pragma unroll
            for (int r = 0; r < 8; ++r) {
                int row = mj * 16 + hi * 8 + r;          // local j
                int col = nk * 16 + n;                   // key index
                float bia = biasb[(size_t)h * (NSEQ * NSEQ) + (size_t)(j0 + row) * NSEQ + col];
                S[row * NSEQ + col] = acc[r] + bia;
            }
        }
    }
    __syncthreads();

    // softmax over 256 keys, 4 lanes per row
    {
        int r = tid >> 2, part = tid & 3;
        float* Sr = S + r * NSEQ + part * 64;
        float mx = -1e30f;
#pragma unroll 8
        for (int u = 0; u < 64; ++u) mx = fmaxf(mx, Sr[u]);
        mx = fmaxf(mx, __shfl_xor(mx, 1, 32));
        mx = fmaxf(mx, __shfl_xor(mx, 2, 32));
        float sum = 0.f;
#pragma unroll 8
        for (int u = 0; u < 64; ++u) { float e = __expf(Sr[u] - mx); Sr[u] = e; sum += e; }
        sum += __shfl_xor(sum, 1, 32);
        sum += __shfl_xor(sum, 2, 32);
        float inv = 1.f / sum;
        bf16* Pr = P + r * NSEQ + part * 64;
#pragma unroll 8
        for (int u = 0; u < 64; ++u) Pr[u] = (bf16)(Sr[u] * inv);
    }
    __syncthreads();

    // O = P x V : 64x32 output, contraction over 256 keys (8 WMMA k-steps)
    {
        int mj = wave & 3, nd = wave >> 2;   // nd in 0..1
        v8f acc = {};
#pragma unroll
        for (int kc = 0; kc < 8; ++kc) {
            v16bf a = load_a(P + mj * 16 * NSEQ + kc * 32, NSEQ, lane);
            v16bf b = load_bT(Vt + nd * 16 * VT_LD + kc * 32, VT_LD, lane);
            acc = WMMA_BF16(a, b, acc);
        }
        int n = lane & 15, hi = lane >> 4;
#pragma unroll
        for (int r = 0; r < 8; ++r) {
            int row = mj * 16 + hi * 8 + r;
            ob[base + (size_t)(j0 + row) * HC + nd * 16 + n] = (bf16)acc[r];
        }
    }
}

// ---------------- Kernel 3: gate * o, then output projection to f32 ---------
__global__ __launch_bounds__(256) void k3_out(
    const bf16* __restrict__ ob, const bf16* __restrict__ gb,
    const float* __restrict__ Wo, float* __restrict__ out)
{
    extern __shared__ char smem[];
    bf16* A  = (bf16*)smem;                      // 64 x 128 gated activations
    bf16* Wt = (bf16*)(smem + 64 * HC * 2);      // 128 x WT_LD transposed Wo

    const int tid  = threadIdx.x;
    const int lane = tid & 31;
    const int wave = tid >> 5;
    const long rows0 = (long)blockIdx.x * 64;

    // gated input tile
    for (int u = 0; u < 32; ++u) {
        int e = u * 256 + tid;                   // 0 .. 8191
        size_t g = (size_t)rows0 * HC + e;
        A[e] = (bf16)((float)gb[g] * (float)ob[g]);
    }
    // Wo transposed into LDS: Wt[cz][hd]
    for (int u = 0; u < 64; ++u) {
        int e = u * 256 + tid;
        int hd = e >> 7, n = e & 127;
        Wt[n * WT_LD + hd] = (bf16)Wo[e];
    }
    __syncthreads();

    int m  = wave & 3;
    int nb = (wave >> 2) * 4;
    v8f acc[4] = {};
#pragma unroll
    for (int kk = 0; kk < 4; ++kk) {
        v16bf a = load_a(A + m * 16 * HC + kk * 32, HC, lane);
#pragma unroll
        for (int t = 0; t < 4; ++t) {
            v16bf b = load_bT(Wt + (nb + t) * 16 * WT_LD + kk * 32, WT_LD, lane);
            acc[t] = WMMA_BF16(a, b, acc[t]);
        }
    }
    int n = lane & 15, hi = lane >> 4;
#pragma unroll
    for (int t = 0; t < 4; ++t) {
#pragma unroll
        for (int r = 0; r < 8; ++r) {
            long row = rows0 + m * 16 + hi * 8 + r;
            out[row * CZ + (nb + t) * 16 + n] = acc[t][r];
        }
    }
}

// ---------------- host launcher ---------------------------------------------
extern "C" void kernel_launch(void* const* d_in, const int* in_sizes, int n_in,
                              void* d_out, int out_size, void* d_ws, size_t ws_size,
                              hipStream_t stream) {
    const float* z    = (const float*)d_in[0];
    const float* ln_g = (const float*)d_in[1];
    const float* ln_b = (const float*)d_in[2];
    const float* Wq   = (const float*)d_in[3];
    const float* Wk   = (const float*)d_in[4];
    const float* Wv   = (const float*)d_in[5];
    const float* Wb   = (const float*)d_in[6];
    const float* Wg   = (const float*)d_in[7];
    const float* Wo   = (const float*)d_in[8];
    float* out = (float*)d_out;

    const size_t ROWS = (size_t)NSEQ * NSEQ;      // 65536
    const size_t MAT  = ROWS * HC;                // 8388608 elements
    char* ws = (char*)d_ws;
    bf16*  qb    = (bf16*)(ws);
    bf16*  kb    = (bf16*)(ws + 1 * MAT * sizeof(bf16));
    bf16*  vb    = (bf16*)(ws + 2 * MAT * sizeof(bf16));
    bf16*  gb    = (bf16*)(ws + 3 * MAT * sizeof(bf16));
    bf16*  obuf  = (bf16*)(ws + 4 * MAT * sizeof(bf16));
    float* biasb = (float*)(ws + 5 * MAT * sizeof(bf16));

    const int SH1 = 64 * CZ * 2 + 128 * WT_LD * 2;                       // ~50 KB
    const int SH2 = 64 * 256 * 4 + 64 * 256 * 2 + 64 * 32 * 2
                  + 2 * 256 * 32 * 2 + 32 * VT_LD * 2;                   // ~149 KB
    const int SH3 = 64 * HC * 2 + 128 * WT_LD * 2;

    (void)hipFuncSetAttribute((const void*)k1_ln_proj,
                              hipFuncAttributeMaxDynamicSharedMemorySize, SH1);
    (void)hipFuncSetAttribute((const void*)k2_attn,
                              hipFuncAttributeMaxDynamicSharedMemorySize, SH2);
    (void)hipFuncSetAttribute((const void*)k3_out,
                              hipFuncAttributeMaxDynamicSharedMemorySize, SH3);

    k1_ln_proj<<<ROWS / 64, 256, SH1, stream>>>(z, ln_g, ln_b, Wq, Wk, Wv, Wb, Wg,
                                                qb, kb, vb, gb, biasb);
    k2_attn<<<dim3(4, 4, NSEQ), 256, SH2, stream>>>(qb, kb, vb, biasb, obuf);
    k3_out<<<ROWS / 64, 256, SH3, stream>>>(obuf, gb, Wo, out);
}
